// GraphMertPairDecoderHead_22024592294148
// MI455X (gfx1250) — matrix operand order repaired and checked by reading, not compiled, not run on hardware
//
#include <hip/hip_runtime.h>

typedef __bf16 bf16_t;
typedef bf16_t bf16x8  __attribute__((ext_vector_type(8)));
typedef bf16_t bf16x16 __attribute__((ext_vector_type(16)));
typedef float  floatx8 __attribute__((ext_vector_type(8)));

#define HID   768
#define TGT   20
#define LBL   30522
#define NPAD  30720   /* LBL padded to multiple of 256 (block N tile) */
#define RTOT  512     /* bs * num_pairs */
#define MROWS 10240   /* RTOT * TGT */

__device__ __forceinline__ bf16x16 cat16(bf16x8 lo, bf16x8 hi) {
  return __builtin_shufflevector(lo, hi, 0,1,2,3,4,5,6,7,8,9,10,11,12,13,14,15);
}

// ---------------------------------------------------------------------------
// Generic bf16 WMMA GEMM: C[M x N] = A[M x K] @ Bt^T (+ bias)
//   A  : row-major bf16, leading dim lda (elements)
//   Bt : [N x K] bf16, K-contiguous (i.e. W transposed), leading dim ldb
//   C  : row-major f32, leading dim ldc
// Block = 256 threads = 8 waves (2 x 4), block tile 64 x 256,
// wave tile 32 x 64 (2 A-frags x 4 B-frags -> 8 WMMAs / 12 b128 loads per k).
// M multiple of 64, K multiple of 32; N handled via Nreal store predication.
// ---------------------------------------------------------------------------
__global__ __launch_bounds__(256)
void wmma_gemm_bf16(const bf16_t* __restrict__ A, int lda,
                    const bf16_t* __restrict__ Bt, int ldb,
                    float* __restrict__ C, int ldc,
                    int Nreal, int K,
                    const float* __restrict__ bias,
                    int nt_store)
{
  const int lane = threadIdx.x & 31;
  const int wid  = threadIdx.x >> 5;
  const int mw   = wid & 1;            // 2 waves along M
  const int nw   = wid >> 1;           // 4 waves along N
  const int m0   = blockIdx.x * 64  + mw * 32;
  const int n0   = blockIdx.y * 256 + nw * 64;
  const int lrow = lane & 15;
  const int lhi  = lane >> 4;

  floatx8 acc[2][4] = {};

  for (int k0 = 0; k0 < K; k0 += 32) {
    bf16x16 afr[2], bfr[4];
    // A fragments (16x32 bf16, ISA 7.12.2 layout):
    //   lanes 0-15: K = k0..k0+7 (V0-3), k0+16..k0+23 (V4-7), row = lane&15
    //   lanes 16-31: K = k0+8..k0+15, k0+24..k0+31, same rows
#pragma unroll
    for (int mi = 0; mi < 2; ++mi) {
      const bf16_t* p = A + (size_t)(m0 + mi * 16 + lrow) * lda + (k0 + lhi * 8);
      bf16x8 lo = *reinterpret_cast<const bf16x8*>(p);
      bf16x8 hi = *reinterpret_cast<const bf16x8*>(p + 16);
      afr[mi] = cat16(lo, hi);
    }
    // B fragments (32x16 bf16): lane&15 = column n; lanes 0-15 hold K=k0..k0+7, k0+16..k0+23,
    // lanes 16-31 hold K=k0+8..k0+15, k0+24..k0+31 (same K order as A).
#pragma unroll
    for (int ni = 0; ni < 4; ++ni) {
      const bf16_t* p = Bt + (size_t)(n0 + ni * 16 + lrow) * ldb + (k0 + lhi * 8);
      bf16x8 lo = *reinterpret_cast<const bf16x8*>(p);
      bf16x8 hi = *reinterpret_cast<const bf16x8*>(p + 16);
      bfr[ni] = cat16(lo, hi);
    }
#pragma unroll
    for (int mi = 0; mi < 2; ++mi)
#pragma unroll
      for (int ni = 0; ni < 4; ++ni)
        acc[mi][ni] = __builtin_amdgcn_wmma_f32_16x16x32_bf16(
            false, afr[mi], false, bfr[ni], (short)0, acc[mi][ni], false, false);
  }

  // D layout: VGPR i -> row = tile_m + i + 8*(lane>=16), col = tile_n + (lane&15)
#pragma unroll
  for (int mi = 0; mi < 2; ++mi) {
#pragma unroll
    for (int ni = 0; ni < 4; ++ni) {
      const int col = n0 + ni * 16 + lrow;
      if (col < Nreal) {
        const float ba = bias ? bias[col] : 0.0f;
        const size_t rbase = (size_t)(m0 + mi * 16 + 8 * lhi);
        if (nt_store) {
#pragma unroll
          for (int i = 0; i < 8; ++i)
            __builtin_nontemporal_store(acc[mi][ni][i] + ba,
                                        &C[(rbase + i) * (size_t)ldc + col]);
        } else {
#pragma unroll
          for (int i = 0; i < 8; ++i)
            C[(rbase + i) * (size_t)ldc + col] = acc[mi][ni][i] + ba;
        }
      }
    }
  }
}

// ---------------------------------------------------------------------------
// Transpose + f32->bf16 convert:  Wt[n*K + k] = (n < N) ? W[k*N + n] : 0
// (pads columns beyond N with zeros)
// ---------------------------------------------------------------------------
__global__ __launch_bounds__(256)
void conv_transpose_kernel(const float* __restrict__ W, int K, int N, int Npad,
                           bf16_t* __restrict__ Wt)
{
  size_t idx = (size_t)blockIdx.x * 256 + threadIdx.x;
  size_t total = (size_t)Npad * (size_t)K;
  if (idx >= total) return;
  int n = (int)(idx / (size_t)K);
  int k = (int)(idx % (size_t)K);
  float v = (n < N) ? W[(size_t)k * N + n] : 0.0f;
  Wt[idx] = (bf16_t)v;
}

// ---------------------------------------------------------------------------
// Gather paired node rows: LR[r][0:768]=nodes[b][l+1], LR[r][768:1536]=nodes[b][r+1]
// ---------------------------------------------------------------------------
__global__ __launch_bounds__(256)
void gather_pairs_kernel(const float* __restrict__ nodes,
                         const int* __restrict__ pairs,
                         bf16_t* __restrict__ LR)
{
  const int r = blockIdx.x;            // 0..511
  const int b = r >> 7;                // / 128
  const int p = r & 127;
  const int li = pairs[((size_t)b * 128 + p) * 2 + 0] + 1;
  const int ri = pairs[((size_t)b * 128 + p) * 2 + 1] + 1;
  const float* ln = nodes + ((size_t)b * 512 + li) * HID;
  const float* rn = nodes + ((size_t)b * 512 + ri) * HID;
  bf16_t* dst = LR + (size_t)r * (2 * HID);
  for (int c = threadIdx.x; c < HID; c += 256) {
    dst[c]       = (bf16_t)ln[c];
    dst[HID + c] = (bf16_t)rn[c];
  }
}

// pos_emb f32 [20][768] -> bf16 [64][768], rows 20..63 zeroed (M padding)
__global__ __launch_bounds__(256)
void conv_pe_kernel(const float* __restrict__ pe, bf16_t* __restrict__ PEc)
{
  size_t idx = (size_t)blockIdx.x * 256 + threadIdx.x;
  if (idx >= (size_t)64 * HID) return;
  int t = (int)(idx / HID);
  PEc[idx] = (t < TGT) ? (bf16_t)pe[idx] : (bf16_t)0.0f;
}

// ---------------------------------------------------------------------------
// Fused gelu(tanh) + LayerNorm, f32 in -> bf16 out. One block per output row.
// If tgt > 0: x = X[row/tgt] + Q[row%tgt] + bias   (broadcast-add structure)
// else:       x = X[row] + bias
// ---------------------------------------------------------------------------
__global__ __launch_bounds__(256)
void act_ln_kernel(const float* __restrict__ X, const float* __restrict__ Q, int tgt,
                   const float* __restrict__ bias, const float* __restrict__ gamma,
                   const float* __restrict__ beta, bf16_t* __restrict__ H)
{
  const int row = blockIdx.x;
  const int tid = threadIdx.x;
  const float* xr;
  const float* qr = nullptr;
  if (tgt > 0) {
    xr = X + (size_t)(row / tgt) * HID;
    qr = Q + (size_t)(row % tgt) * HID;
  } else {
    xr = X + (size_t)row * HID;
  }

  float v[3];
  float s = 0.0f, ss = 0.0f;
#pragma unroll
  for (int j = 0; j < 3; ++j) {
    const int c = tid + j * 256;
    float x = xr[c] + bias[c];
    if (qr) x += qr[c];
    // jax.nn.gelu (approximate=True, tanh form)
    const float u = 0.7978845608028654f * (x + 0.044715f * x * x * x);
    const float g = 0.5f * x * (1.0f + tanhf(u));
    v[j] = g;
    s += g;
    ss += g * g;
  }
#pragma unroll
  for (int off = 16; off > 0; off >>= 1) {
    s  += __shfl_xor(s, off, 32);
    ss += __shfl_xor(ss, off, 32);
  }
  __shared__ float sred[8], ssred[8];
  const int wid = tid >> 5, lane = tid & 31;
  if (lane == 0) { sred[wid] = s; ssred[wid] = ss; }
  __syncthreads();
  if (wid == 0) {
    float a  = (lane < 8) ? sred[lane]  : 0.0f;
    float b2 = (lane < 8) ? ssred[lane] : 0.0f;
#pragma unroll
    for (int off = 4; off > 0; off >>= 1) {
      a  += __shfl_xor(a, off, 32);
      b2 += __shfl_xor(b2, off, 32);
    }
    if (lane == 0) { sred[0] = a; ssred[0] = b2; }
  }
  __syncthreads();
  const float mean = sred[0] * (1.0f / HID);
  const float var  = ssred[0] * (1.0f / HID) - mean * mean;
  const float rs   = rsqrtf(var + 1e-5f);
#pragma unroll
  for (int j = 0; j < 3; ++j) {
    const int c = tid + j * 256;
    const float y = (v[j] - mean) * rs * gamma[c] + beta[c];
    H[(size_t)row * HID + c] = (bf16_t)y;
  }
}

// ---------------------------------------------------------------------------
extern "C" void kernel_launch(void* const* d_in, const int* in_sizes, int n_in,
                              void* d_out, int out_size, void* d_ws, size_t ws_size,
                              hipStream_t stream)
{
  const float* nodes = (const float*)d_in[0];   // [4,512,768]
  const int*   pairs = (const int*)d_in[1];     // [4,128,2]
  const float* pos   = (const float*)d_in[2];   // [20,768]
  const float* W1    = (const float*)d_in[3];   // [2304,768]
  const float* b1    = (const float*)d_in[4];
  const float* g1    = (const float*)d_in[5];
  const float* be1   = (const float*)d_in[6];
  const float* W2    = (const float*)d_in[7];   // [768,768]
  const float* b2    = (const float*)d_in[8];
  const float* g2    = (const float*)d_in[9];
  const float* be2   = (const float*)d_in[10];
  const float* Wd    = (const float*)d_in[11];  // [768,30522]
  const float* bd    = (const float*)d_in[12];
  float* out = (float*)d_out;                   // [10240, 30522]

  // workspace carve-out (256-B aligned chunks)
  char* w = (char*)d_ws;
  auto carve = [&](size_t bytes) {
    void* p = (void*)w;
    w += (bytes + 255) & ~(size_t)255;
    return p;
  };
  bf16_t* W1t = (bf16_t*)carve((size_t)HID * 2304 * 2);         // [768][2304]
  bf16_t* W2t = (bf16_t*)carve((size_t)HID * HID * 2);          // [768][768]
  bf16_t* Wdt = (bf16_t*)carve((size_t)NPAD * HID * 2);         // [30720][768]
  bf16_t* LR  = (bf16_t*)carve((size_t)RTOT * 2 * HID * 2);     // [512][1536]
  bf16_t* PEc = (bf16_t*)carve((size_t)64 * HID * 2);           // [64][768]
  float*  P   = (float*)carve((size_t)RTOT * HID * 4);          // [512][768]
  float*  Qm  = (float*)carve((size_t)64 * HID * 4);            // [64][768]
  bf16_t* h1  = (bf16_t*)carve((size_t)MROWS * HID * 2);        // [10240][768]
  bf16_t* h2  = (bf16_t*)carve((size_t)MROWS * HID * 2);        // [10240][768]
  float*  G2  = out;  // gemm2 pre-activation scratch; fully overwritten by gemm3

  // 1) weight convert + transpose (cheap one-time prep)
  {
    size_t t1 = (size_t)HID * 2304;
    conv_transpose_kernel<<<(unsigned)((t1 + 255) / 256), 256, 0, stream>>>(W1, 2304, HID, HID, W1t);
    size_t t2 = (size_t)HID * HID;
    conv_transpose_kernel<<<(unsigned)((t2 + 255) / 256), 256, 0, stream>>>(W2, HID, HID, HID, W2t);
    size_t t3 = (size_t)NPAD * HID;
    conv_transpose_kernel<<<(unsigned)((t3 + 255) / 256), 256, 0, stream>>>(Wd, HID, LBL, NPAD, Wdt);
  }

  // 2) gather pair rows + convert pos_emb
  gather_pairs_kernel<<<RTOT, 256, 0, stream>>>(nodes, pairs, LR);
  conv_pe_kernel<<<(64 * HID + 255) / 256, 256, 0, stream>>>(pos, PEc);

  // 3) structured GEMM1: P = [lh|rh] @ W1[0:1536,:]   (M=512, K=1536, N=768)
  wmma_gemm_bf16<<<dim3(RTOT / 64, HID / 256), 256, 0, stream>>>(
      LR, 2 * HID, W1t, 2304, P, HID, HID, 2 * HID, nullptr, 0);
  //    Q = pe @ W1[1536:,:]                            (M=64 padded, K=768, N=768)
  wmma_gemm_bf16<<<dim3(1, HID / 256), 256, 0, stream>>>(
      PEc, HID, W1t + 1536, 2304, Qm, HID, HID, HID, nullptr, 0);

  // 4) h1 = LN(gelu(P[r] + Q[t] + b1))
  act_ln_kernel<<<MROWS, 256, 0, stream>>>(P, Qm, TGT, b1, g1, be1, h1);

  // 5) GEMM2 (M=10240, K=768, N=768) -> f32 scratch in d_out (re-read: cached stores)
  wmma_gemm_bf16<<<dim3(MROWS / 64, HID / 256), 256, 0, stream>>>(
      h1, HID, W2t, HID, G2, HID, HID, HID, nullptr, 0);

  // 6) h2 = LN(gelu(G2 + b2))
  act_ln_kernel<<<MROWS, 256, 0, stream>>>(G2, nullptr, 0, b2, g2, be2, h2);

  // 7) GEMM3 (M=10240, K=768, N=30522 padded 30720) + bd -> d_out
  //    Output is write-once/stream -> non-temporal stores keep L2 for Wd^T + h2.
  wmma_gemm_bf16<<<dim3(MROWS / 64, NPAD / 256), 256, 0, stream>>>(
      h2, HID, Wdt, HID, out, LBL, LBL, HID, bd, 1);
}
